// GraftNet_13718125543550
// MI455X (gfx1250) — matrix operand order repaired and checked by reading, not compiled, not run on hardware
//
#include <hip/hip_runtime.h>
#include <hip/hip_bf16.h>
#include <math.h>

// ---------------------------------------------------------------------------
// GraftNet forward for MI455X (gfx1250, wave32, WMMA).
// HBM-bound (~250 GFLOP vs ~1-2 GB traffic @ 23.3 TB/s): all dense projections
// run on v_wmma_f32_16x16x32_f16; attention is fused so the (B,F,D) "att"
// tensor (134 MB) is never materialized. GEMM hot loop is branch-free:
// gather path is a template parameter, K-tail is peeled into one guarded step.
// ---------------------------------------------------------------------------

#define B_ 16
#define E_ 2048
#define F_ 8192
#define Q_ 20
#define D_ 256
#define WD_ 300
#define NWORD 50000
#define VERY_NEG_ (-1.0e11f)
#define VERY_SMALL_ (1e-10f)
#define PR_LAMBDA_ 0.8f
#define FACT_SCALE_ 3.0f

typedef __attribute__((ext_vector_type(16))) _Float16 v16h;
typedef __attribute__((ext_vector_type(8)))  float    v8f;

// ---------------------------------------------------------------------------
// WMMA GEMM: Y[M,N] = act( src[M,K] @ W[N,K]^T + bias (+ bias2) )
// src row m = GATHER ? Emb[idx[m], :] : X[m, :]
// Block = 256 threads = 8 waves. Each wave computes a 16x32 output tile
// (two WMMA accumulators sharing one A fragment); block = 16(M) x 256(N).
// A tile (16x32 fp32) staged cooperatively in LDS once per K step.
// ---------------------------------------------------------------------------
template<bool GATHER>
__global__ __launch_bounds__(256)
void gn_wmma_gemm(const float* __restrict__ X, const int* __restrict__ idx,
                  const float* __restrict__ Emb,
                  const float* __restrict__ W,
                  const float* __restrict__ bias, const float* __restrict__ bias2,
                  float* __restrict__ Y, int M, int N, int K, int act)
{
    __shared__ float lA[16 * 32];
    const int tid      = threadIdx.x;
    const int wave     = tid >> 5;
    const int lane     = tid & 31;
    const int laneM    = lane & 15;
    const int laneHalf = lane >> 4;
    const int tileM    = (int)blockIdx.y << 4;
    const int tileN    = ((int)blockIdx.x << 8) + (wave << 5);

    // A staging assignment: thread t stages row ar = t/16, 2 floats at ak.
    const int ar = tid >> 4;
    const int ak = (tid & 15) << 1;
    const float* abase;
    if (GATHER) abase = Emb + (size_t)idx[tileM + ar] * K + ak;
    else        abase = X   + (size_t)(tileM + ar)    * K + ak;

    // B: each lane owns K 8h..8h+7 and 16+8h..16+8h+7 (h = lane/16) of rows n0 and n1.
    const int n0 = tileN + laneM;
    const int n1 = n0 + 16;
    const float* brow0 = W + (size_t)(n0 < N ? n0 : 0) * K + (laneHalf << 3);
    const float* brow1 = W + (size_t)(n1 < N ? n1 : 0) * K + (laneHalf << 3);

    v8f acc0 = {0.f, 0.f, 0.f, 0.f, 0.f, 0.f, 0.f, 0.f};
    v8f acc1 = {0.f, 0.f, 0.f, 0.f, 0.f, 0.f, 0.f, 0.f};

    int kk = 0;
    // ---- main loop: no bounds checks, vectorized loads ----
    for (; kk + 32 <= K; kk += 32) {
        float2 av = *(const float2*)(abase + kk);
        *(float2*)&lA[(ar << 5) + ak] = av;
        __syncthreads();

        v16h a, b0, b1;
#pragma unroll
        for (int i = 0; i < 8; ++i) {
            // A f16 layout: lanes 0-15 hold K 0..7 & 16..23, lanes 16-31 +8
            a[i]     = (_Float16)lA[(laneM << 5) + (laneHalf << 3) + i];
            a[i + 8] = (_Float16)lA[(laneM << 5) + 16 + (laneHalf << 3) + i];
        }
#pragma unroll
        for (int i = 0; i < 16; ++i) {
            // B f16 layout: lanes 0-15 hold K 0..7 & 16..23, lanes 16-31 +8
            b0[i] = (_Float16)brow0[kk + i + ((i >> 3) << 3)];
            b1[i] = (_Float16)brow1[kk + i + ((i >> 3) << 3)];
        }
        acc0 = __builtin_amdgcn_wmma_f32_16x16x32_f16(false, a, false, b0,
                                                      (short)0, acc0, false, false);
        acc1 = __builtin_amdgcn_wmma_f32_16x16x32_f16(false, a, false, b1,
                                                      (short)0, acc1, false, false);
        __syncthreads();
    }

    // ---- peeled tail (K % 32 != 0): guarded, runs at most once ----
    if (kk < K) {
        float2 av;
        av.x = (kk + ak     < K) ? abase[kk]     : 0.f;
        av.y = (kk + ak + 1 < K) ? abase[kk + 1] : 0.f;
        *(float2*)&lA[(ar << 5) + ak] = av;
        __syncthreads();

        v16h a, b0, b1;
#pragma unroll
        for (int i = 0; i < 8; ++i) {
            a[i]     = (_Float16)lA[(laneM << 5) + (laneHalf << 3) + i];
            a[i + 8] = (_Float16)lA[(laneM << 5) + 16 + (laneHalf << 3) + i];
        }
#pragma unroll
        for (int i = 0; i < 16; ++i) {
            int kb = kk + (laneHalf << 3) + i + ((i >> 3) << 3);
            b0[i] = (kb < K) ? (_Float16)brow0[kk + i + ((i >> 3) << 3)] : (_Float16)0.f;
            b1[i] = (kb < K) ? (_Float16)brow1[kk + i + ((i >> 3) << 3)] : (_Float16)0.f;
        }
        acc0 = __builtin_amdgcn_wmma_f32_16x16x32_f16(false, a, false, b0,
                                                      (short)0, acc0, false, false);
        acc1 = __builtin_amdgcn_wmma_f32_16x16x32_f16(false, a, false, b1,
                                                      (short)0, acc1, false, false);
    }

    // ---- store: C/D layout m = vgpr + 8*(lane/16), n = lane%16 ----
    float bv0 = 0.f, bv1 = 0.f;
    if (n0 < N) bv0 = (bias ? bias[n0] : 0.f) + (bias2 ? bias2[n0] : 0.f);
    if (n1 < N) bv1 = (bias ? bias[n1] : 0.f) + (bias2 ? bias2[n1] : 0.f);
#pragma unroll
    for (int r = 0; r < 8; ++r) {
        int m = tileM + (laneHalf << 3) + r;
        float v0 = acc0[r] + bv0;
        float v1 = acc1[r] + bv1;
        if (act == 1) { v0 = fmaxf(v0, 0.f); v1 = fmaxf(v1, 0.f); }
        if (n0 < N) Y[(size_t)m * N + n0] = v0;
        if (n1 < N) Y[(size_t)m * N + n1] = v1;
    }
}

// ---------------------------------------------------------------------------
// LSTM scan: one workgroup, 1024 threads. h shared in LDS (16 KB), c in regs.
// ---------------------------------------------------------------------------
__global__ __launch_bounds__(1024)
void gn_lstm_scan(const float* __restrict__ gx, const float* __restrict__ w_hh,
                  float* __restrict__ qh, float* __restrict__ qnode)
{
    __shared__ float hS[B_ * D_];
    const int tid   = threadIdx.x;
    const int d     = tid & (D_ - 1);
    const int gbase = (tid >> 8) << 2;
    float c[4] = {0.f, 0.f, 0.f, 0.f};
    for (int e = tid; e < B_ * D_; e += 1024) hS[e] = 0.f;
    __syncthreads();

    for (int step = 0; step < Q_; ++step) {
        float z[4][4];
        for (int bi = 0; bi < 4; ++bi) {
            int b = gbase + bi;
            const float* hrow = hS + b * D_;
            const float* gxr  = gx + (size_t)(b * Q_ + step) * (4 * D_);
            for (int g = 0; g < 4; ++g) {
                const float* wr = w_hh + (size_t)(g * D_ + d) * D_;
                float a = gxr[g * D_ + d];
                for (int k = 0; k < D_; ++k) a += hrow[k] * wr[k];
                z[bi][g] = a;
            }
        }
        __syncthreads();
        for (int bi = 0; bi < 4; ++bi) {
            int b = gbase + bi;
            float ig = 1.f / (1.f + expf(-z[bi][0]));
            float fg = 1.f / (1.f + expf(-z[bi][1]));
            float gg = tanhf(z[bi][2]);
            float og = 1.f / (1.f + expf(-z[bi][3]));
            c[bi] = fg * c[bi] + ig * gg;
            float h = og * tanhf(c[bi]);
            hS[b * D_ + d] = h;
            qh[(size_t)(b * Q_ + step) * D_ + d] = h;
            if (step == Q_ - 1) qnode[b * D_ + d] = h;
        }
        __syncthreads();
    }
}

// ---------------------------------------------------------------------------
// Fused attention: per (b,f): dots over q, masked softmax over q,
// Wf[b,f] = sum_q p_q * dot_q / 16.  "att" tensor never materialized.
// ---------------------------------------------------------------------------
__global__ __launch_bounds__(256)
void gn_attention_wf(const float* __restrict__ qh, const float* __restrict__ lf,
                     const int* __restrict__ query_text, float* __restrict__ Wf)
{
    __shared__ float qS[Q_ * D_];
    __shared__ float mS[Q_];
    const int b   = blockIdx.y;
    const int tid = threadIdx.x;
    const int f   = blockIdx.x * 256 + tid;
    for (int e = tid; e < Q_ * D_; e += 256) qS[e] = qh[(size_t)b * Q_ * D_ + e];
    if (tid < Q_) mS[tid] = (query_text[b * Q_ + tid] != NWORD) ? 0.f : VERY_NEG_;
    __syncthreads();

    const float* lfr = lf + ((size_t)b * F_ + f) * D_;
    float dot[Q_];
#pragma unroll
    for (int q = 0; q < Q_; ++q) dot[q] = 0.f;
    for (int k = 0; k < D_; ++k) {
        float lv = lfr[k];
#pragma unroll
        for (int q = 0; q < Q_; ++q) dot[q] += qS[q * D_ + k] * lv;
    }
    float mx = -3.4e38f;
    float logit[Q_];
#pragma unroll
    for (int q = 0; q < Q_; ++q) {
        logit[q] = dot[q] * (1.f / 16.f) + mS[q];
        mx = fmaxf(mx, logit[q]);
    }
    float sum = 0.f, wf = 0.f;
#pragma unroll
    for (int q = 0; q < Q_; ++q) {
        float p = expf(logit[q] - mx);
        sum += p;
        wf  += p * dot[q];
    }
    Wf[(size_t)b * F_ + f] = wf * (1.f / 16.f) / sum;
}

__global__ __launch_bounds__(256)
void gn_rowmax(const float* __restrict__ Wf, float* __restrict__ mx)
{
    __shared__ float red[256];
    int b = blockIdx.x, tid = threadIdx.x;
    float m = -3.4e38f;
    for (int f = tid; f < F_; f += 256) m = fmaxf(m, Wf[(size_t)b * F_ + f]);
    red[tid] = m;
    __syncthreads();
    for (int s = 128; s > 0; s >>= 1) {
        if (tid < s) red[tid] = fmaxf(red[tid], red[tid + s]);
        __syncthreads();
    }
    if (tid == 0) mx[b] = red[0];
}

__global__ void gn_wtexp_scatter(float* __restrict__ Wt, const float* __restrict__ mx,
                                 const int* __restrict__ e2f_ent, float* __restrict__ e2fs)
{
    long long i = (long long)blockIdx.x * blockDim.x + threadIdx.x;
    if (i >= (long long)B_ * F_) return;
    int b = (int)(i / F_);
    float w = expf(Wt[i] - mx[b]);
    Wt[i] = w;
    atomicAdd(&e2fs[(size_t)b * E_ + e2f_ent[i]], w);
}

// ---------------------------------------------------------------------------
// elementwise / scatter glue
// ---------------------------------------------------------------------------
__global__ void gn_zero(float* __restrict__ p, long long n)
{
    long long i = (long long)blockIdx.x * blockDim.x + threadIdx.x;
    if (i < n) p[i] = 0.f;
}

__global__ void gn_copy(float* __restrict__ dst, const float* __restrict__ src, long long n)
{
    long long i = (long long)blockIdx.x * blockDim.x + threadIdx.x;
    if (i < n) dst[i] = src[i];
}

__global__ void gn_norm(const float* __restrict__ Wt, const float* __restrict__ pr,
                        const float* __restrict__ e2fs, const int* __restrict__ e2f_ent,
                        float* __restrict__ nrm)
{
    long long i = (long long)blockIdx.x * blockDim.x + threadIdx.x;
    if (i >= (long long)B_ * F_) return;
    int b = (int)(i / F_);
    int e = e2f_ent[i];
    nrm[i] = Wt[i] * pr[b * E_ + e] / fmaxf(e2fs[b * E_ + e], VERY_SMALL_);
}

__global__ void gn_e2f_build(float* __restrict__ e2f, const float* __restrict__ head,
                             const float* __restrict__ nrm, const int* __restrict__ e2f_ent)
{
    long long i = (long long)blockIdx.x * blockDim.x + threadIdx.x;
    if (i >= (long long)B_ * F_ * D_) return;
    long long bf = i / D_;
    int d = (int)(i % D_);
    int b = (int)(bf / F_);
    int e = e2f_ent[bf];
    float v = e2f[i] + head[((size_t)b * E_ + e) * D_ + d];
    e2f[i] = fmaxf(v, 0.f) * nrm[bf];
}

__global__ void gn_scatter_tail(const float* __restrict__ tail, float* __restrict__ f2e,
                                const int* __restrict__ f2e_ent)
{
    long long i = (long long)blockIdx.x * blockDim.x + threadIdx.x;
    if (i >= (long long)B_ * F_ * D_) return;
    long long bf = i / D_;
    int d = (int)(i % D_);
    int b = (int)(bf / F_);
    int e = f2e_ent[bf];
    atomicAdd(&f2e[((size_t)b * E_ + e) * D_ + d], tail[i]);
}

__global__ void gn_scatter_pr(const float* __restrict__ nrm, float* __restrict__ prs,
                              const int* __restrict__ f2e_ent)
{
    long long i = (long long)blockIdx.x * blockDim.x + threadIdx.x;
    if (i >= (long long)B_ * F_) return;
    int b = (int)(i / F_);
    atomicAdd(&prs[b * E_ + f2e_ent[i]], nrm[i]);
}

__global__ void gn_pr_combine(float* __restrict__ pr, const float* __restrict__ prs)
{
    long long i = (long long)blockIdx.x * blockDim.x + threadIdx.x;
    if (i >= (long long)B_ * E_) return;
    pr[i] = PR_LAMBDA_ * prs[i] + (1.f - PR_LAMBDA_) * pr[i];
}

__global__ void gn_nxt_build(const float* __restrict__ le, const float* __restrict__ q2ev,
                             const float* __restrict__ f2e, float* __restrict__ nxt)
{
    long long i = (long long)blockIdx.x * blockDim.x + threadIdx.x;
    if (i >= (long long)B_ * E_ * D_) return;
    long long be = i / D_;
    int d = (int)(i % D_);
    int b = (int)(be / E_);
    size_t base = (size_t)be * (3 * D_);
    nxt[base + d]          = le[i];
    nxt[base + D_ + d]     = q2ev[b * D_ + d];
    nxt[base + 2 * D_ + d] = FACT_SCALE_ * fmaxf(f2e[i], 0.f);
}

__global__ __launch_bounds__(256)
void gn_qnode_update(const float* __restrict__ pr, const float* __restrict__ T,
                     float* __restrict__ qn)
{
    int b = blockIdx.x, d = threadIdx.x;
    float acc = 0.f;
    for (int e = 0; e < E_; ++e)
        acc += pr[b * E_ + e] * T[((size_t)b * E_ + e) * D_ + d];
    qn[b * D_ + d] = acc;
}

__global__ void gn_score(const float* __restrict__ le, const float* __restrict__ sw,
                         const float* __restrict__ sb, float* __restrict__ out)
{
    long long i = (long long)blockIdx.x * blockDim.x + threadIdx.x;
    if (i >= (long long)B_ * E_) return;
    const float* r = le + (size_t)i * D_;
    float a = sb[0];
    for (int k = 0; k < D_; ++k) a += r[k] * sw[k];
    out[i] = a;
}

// ---------------------------------------------------------------------------
// host orchestration
// ---------------------------------------------------------------------------
static inline void launch_gemm(hipStream_t s, const float* X, const int* idx,
                               const float* Emb, const float* W,
                               const float* bias, const float* bias2,
                               float* Y, int M, int N, int K, int act)
{
    dim3 grid((N + 255) / 256, M / 16);
    if (idx)
        gn_wmma_gemm<true><<<grid, 256, 0, s>>>(X, idx, Emb, W, bias, bias2, Y, M, N, K, act);
    else
        gn_wmma_gemm<false><<<grid, 256, 0, s>>>(X, idx, Emb, W, bias, bias2, Y, M, N, K, act);
}

static inline int nb(long long n) { return (int)((n + 255) / 256); }

extern "C" void kernel_launch(void* const* d_in, const int* in_sizes, int n_in,
                              void* d_out, int out_size, void* d_ws, size_t ws_size,
                              hipStream_t stream)
{
    (void)in_sizes; (void)n_in; (void)out_size; (void)ws_size;
    const int*   local_entity   = (const int*)  d_in[0];
    const float* q2e_adj        = (const float*)d_in[1];
    const int*   kb_fact_rel    = (const int*)  d_in[2];
    const int*   query_text     = (const int*)  d_in[3];
    /* answer_dist d_in[4] unused */
    const int*   e2f_ent        = (const int*)  d_in[5];
    const int*   f2e_ent        = (const int*)  d_in[6];
    const float* word_emb       = (const float*)d_in[7];
    const float* entity_emb     = (const float*)d_in[8];
    const float* relation_emb   = (const float*)d_in[9];
    const float* entity_lin_w   = (const float*)d_in[10];
    const float* entity_lin_b   = (const float*)d_in[11];
    const float* relation_lin_w = (const float*)d_in[12];
    const float* relation_lin_b = (const float*)d_in[13];
    const float* lstm_w_ih      = (const float*)d_in[14];
    const float* lstm_w_hh      = (const float*)d_in[15];
    const float* lstm_b_ih      = (const float*)d_in[16];
    const float* lstm_b_hh      = (const float*)d_in[17];
    const float* q2e_w          = (const float*)d_in[18];
    const float* q2e_b          = (const float*)d_in[19];
    const float* e2q_w          = (const float*)d_in[20];
    const float* e2q_b          = (const float*)d_in[21];
    const float* e2e_w          = (const float*)d_in[22];
    const float* e2e_b          = (const float*)d_in[23];
    const float* kb_head_w      = (const float*)d_in[24];
    const float* kb_head_b      = (const float*)d_in[25];
    const float* kb_tail_w      = (const float*)d_in[26];
    const float* kb_tail_b      = (const float*)d_in[27];
    const float* kb_self_w      = (const float*)d_in[28];
    const float* kb_self_b      = (const float*)d_in[29];
    const float* score_w        = (const float*)d_in[30];
    const float* score_b        = (const float*)d_in[31];
    float* out = (float*)d_out;

    size_t off = 0;
    char* base = (char*)d_ws;
    auto alloc = [&](size_t nfloats) -> float* {
        float* p = (float*)(base + off);
        off += nfloats * sizeof(float);
        off = (off + 255) & ~(size_t)255;
        return p;
    };

    const long long BF  = (long long)B_ * F_;
    const long long BE  = (long long)B_ * E_;
    const long long BFD = BF * D_;
    const long long BED = BE * D_;

    float* GX    = alloc((size_t)B_ * Q_ * 4 * D_);
    float* QH    = alloc((size_t)B_ * Q_ * D_);
    float* QN    = alloc((size_t)B_ * D_);
    float* Q2EV  = alloc((size_t)B_ * D_);
    float* LF    = alloc((size_t)BFD);
    float* LE    = alloc((size_t)BED);
    float* WT    = alloc((size_t)BF);
    float* MAXB  = alloc(16);
    float* E2FS  = alloc((size_t)BE);
    float* PR    = alloc((size_t)BE);
    float* PRS   = alloc((size_t)BE);
    float* NORM  = alloc((size_t)BF);
    float* E2F   = alloc((size_t)BFD);   // fact_self -> e2f_emb (in place)
    float* HEADT = alloc((size_t)BED);   // head, later reused for t = nxt@e2q
    float* TAIL  = alloc((size_t)BFD);
    float* F2E   = alloc((size_t)BED);   // ent_self + scattered tail
    float* NXT   = alloc((size_t)BE * 3 * D_);

    // ---- query encoder -----------------------------------------------------
    launch_gemm(stream, nullptr, query_text, word_emb, lstm_w_ih,
                lstm_b_ih, lstm_b_hh, GX, B_ * Q_, 4 * D_, WD_, 0);
    gn_lstm_scan<<<1, 1024, 0, stream>>>(GX, lstm_w_hh, QH, QN);

    // ---- fact / entity projections (fused gather + GEMM) -------------------
    launch_gemm(stream, nullptr, kb_fact_rel, relation_emb, relation_lin_w,
                relation_lin_b, nullptr, LF, (int)BF, D_, 2 * WD_, 0);
    launch_gemm(stream, nullptr, local_entity, entity_emb, entity_lin_w,
                entity_lin_b, nullptr, LE, (int)BE, D_, WD_, 0);

    // ---- fused attention -> Wf, then W_tilde + e2f_softmax -----------------
    gn_zero<<<nb(BE), 256, 0, stream>>>(E2FS, BE);
    {
        dim3 g(F_ / 256, B_);
        gn_attention_wf<<<g, 256, 0, stream>>>(QH, LF, query_text, WT);
    }
    gn_rowmax<<<B_, 256, 0, stream>>>(WT, MAXB);
    gn_wtexp_scatter<<<nb(BF), 256, 0, stream>>>(WT, MAXB, e2f_ent, E2FS);

    gn_copy<<<nb(BE), 256, 0, stream>>>(PR, q2e_adj, BE);

    // ---- GNN layers --------------------------------------------------------
    for (int i = 0; i < 3; ++i) {
        const float* self_w = kb_self_w + (size_t)i * D_ * D_;
        const float* self_b = kb_self_b + (size_t)i * D_;

        launch_gemm(stream, QN, nullptr, nullptr, q2e_w + (size_t)i * D_ * D_,
                    q2e_b + (size_t)i * D_, nullptr, Q2EV, B_, D_, D_, 0);
        launch_gemm(stream, LF, nullptr, nullptr, self_w, self_b, nullptr,
                    E2F, (int)BF, D_, D_, 0);
        launch_gemm(stream, LE, nullptr, nullptr, kb_head_w + (size_t)i * D_ * D_,
                    kb_head_b + (size_t)i * D_, nullptr, HEADT, (int)BE, D_, D_, 0);

        gn_norm<<<nb(BF), 256, 0, stream>>>(WT, PR, E2FS, e2f_ent, NORM);
        gn_e2f_build<<<nb(BFD), 256, 0, stream>>>(E2F, HEADT, NORM, e2f_ent);

        launch_gemm(stream, E2F, nullptr, nullptr, kb_tail_w + (size_t)i * D_ * D_,
                    kb_tail_b + (size_t)i * D_, nullptr, TAIL, (int)BF, D_, D_, 0);
        launch_gemm(stream, LE, nullptr, nullptr, self_w, self_b, nullptr,
                    F2E, (int)BE, D_, D_, 0);
        gn_scatter_tail<<<nb(BFD), 256, 0, stream>>>(TAIL, F2E, f2e_ent);

        gn_zero<<<nb(BE), 256, 0, stream>>>(PRS, BE);
        gn_scatter_pr<<<nb(BF), 256, 0, stream>>>(NORM, PRS, f2e_ent);
        gn_pr_combine<<<nb(BE), 256, 0, stream>>>(PR, PRS);

        gn_nxt_build<<<nb(BED), 256, 0, stream>>>(LE, Q2EV, F2E, NXT);

        launch_gemm(stream, NXT, nullptr, nullptr, e2q_w + (size_t)i * D_ * 3 * D_,
                    e2q_b + (size_t)i * D_, nullptr, HEADT, (int)BE, D_, 3 * D_, 0);
        gn_qnode_update<<<B_, 256, 0, stream>>>(PR, HEADT, QN);

        launch_gemm(stream, NXT, nullptr, nullptr, e2e_w + (size_t)i * D_ * 3 * D_,
                    e2e_b + (size_t)i * D_, nullptr, LE, (int)BE, D_, 3 * D_, 1);
    }

    // ---- score -------------------------------------------------------------
    gn_score<<<nb(BE), 256, 0, stream>>>(LE, score_w, score_b, out);
}
